// GRU_49297634623758
// MI455X (gfx1250) — compile-verified, matrix operand order checked by reading, not run on hardware
//
#include <hip/hip_runtime.h>
#include <hip/hip_fp16.h>

// ---------------------------------------------------------------------------
// Bidirectional GRU tagger for MI455X (gfx1250, wave32).
//   Phase 1: embedding gather -> f16, weight conversion -> f16
//   Phase 2: input projections gi = x @ W_ih.T + b_ih  (v_wmma_f32_16x16x32_f16,
//            B tiles staged via async global->LDS double buffering)
//   Phase 3: persistent recurrence, W_hh staged into LDS by the Tensor Data
//            Mover (tensor_load_to_lds + s_wait_tensorcnt); 16 WGs/direction,
//            h exchanged through L2 with one release/acquire sync per step
//   Phase 4: output projection out = h @ W_out.T       (WMMA)
// ---------------------------------------------------------------------------

typedef __attribute__((ext_vector_type(16))) _Float16 v16h;
typedef __attribute__((ext_vector_type(8)))  _Float16 v8h;
typedef __attribute__((ext_vector_type(2)))  _Float16 f16x2;
typedef __attribute__((ext_vector_type(8)))  float    v8f;
typedef __attribute__((ext_vector_type(4)))  unsigned int v4u;
typedef __attribute__((ext_vector_type(8)))  int      v8i;
typedef __attribute__((ext_vector_type(4)))  int      v4i;
typedef int g_v4i __attribute__((vector_size(16)));   // async-LDS builtin pointee

// Pointer casts for the async global->LDS builtin: (v4i AS1*, v4i AS3*, imm, imm)
#define GPTR(p) ((__attribute__((address_space(1))) g_v4i*)(void*)(p))
#define LPTR(p) ((__attribute__((address_space(3))) g_v4i*)(p))

#define T_LEN 4096
#define E_DIM 1024
#define H_DIM 512
#define G3H   1536
#define NWG   16      // workgroups per direction in the recurrence

#if defined(__AMDGCN__) && __has_builtin(__builtin_amdgcn_global_load_async_to_lds_b128) && __has_builtin(__builtin_amdgcn_s_wait_asynccnt)
#define HAVE_ASYNC_LDS 1
#else
#define HAVE_ASYNC_LDS 0
#endif

#if defined(__AMDGCN__) && __has_builtin(__builtin_amdgcn_tensor_load_to_lds) && __has_builtin(__builtin_amdgcn_s_wait_tensorcnt)
#define HAVE_TDM 1
#else
#define HAVE_TDM 0
#endif

// ---------------------------------------------------------------------------
// Embedding gather + f32->f16 convert. Also zeroes the sync counters.
// ---------------------------------------------------------------------------
__global__ __launch_bounds__(256) void k_gather(
    const int* __restrict__ sent, const float* __restrict__ emb,
    _Float16* __restrict__ x16, unsigned* __restrict__ sync)
{
  if (blockIdx.x == 0 && threadIdx.x < 2) sync[threadIdx.x] = 0u;
  size_t i = ((size_t)blockIdx.x * blockDim.x + threadIdx.x) * 2;
  int t = (int)(i >> 10);
  int e = (int)(i & 1023);
  const float2 v = *(const float2*)(emb + (size_t)sent[t] * E_DIM + e);
  f16x2 o; o.x = (_Float16)v.x; o.y = (_Float16)v.y;
  *(f16x2*)(x16 + i) = o;
}

// Generic f32 -> f16 conversion (n must be even).
__global__ __launch_bounds__(256) void k_cvt(
    const float* __restrict__ in, _Float16* __restrict__ out, int n)
{
  int i = ((int)blockIdx.x * blockDim.x + threadIdx.x) * 2;
  if (i < n) {
    f16x2 o; o.x = (_Float16)in[i]; o.y = (_Float16)in[i + 1];
    *(f16x2*)(out + i) = o;
  }
}

// ---------------------------------------------------------------------------
// WMMA GEMM: C[M][N] = A[M][K] * B[N][K]^T + bias[N]
// A, B row-major f16; per-wave tile 16 x (16*NTILES); block = 8 waves = 128 rows.
// Operand layout per CDNA5 ISA 7.12.2 (lane l: row/col l%16, K chunks at
// 8*(l/16) and 16+8*(l/16)).  B tiles shared by the 8 waves are staged in LDS
// via async global->LDS loads (double-buffered); A is register-pipelined one
// k-iteration ahead.  All B ds_loads of a k-step are issued before the WMMAs
// so the scheduler can overlap LDS latency with matrix math.
// ---------------------------------------------------------------------------
template <int NTILES>
__global__ __launch_bounds__(256) void k_wmma_gemm(
    const _Float16* __restrict__ A, const _Float16* __restrict__ B,
    const float* __restrict__ bias, float* __restrict__ C,
    int M, int N, int K)
{
  const int tid   = threadIdx.x;
  const int wave  = tid >> 5;
  const int lane  = tid & 31;
  const int lmod  = lane & 15;
  const int lhalf = lane >> 4;
  const int m0 = blockIdx.y * 128 + wave * 16;
  const int n0 = blockIdx.x * (16 * NTILES);

  v8f acc[NTILES];
#pragma unroll
  for (int i = 0; i < NTILES; ++i) acc[i] = (v8f){0.f,0.f,0.f,0.f,0.f,0.f,0.f,0.f};

#if HAVE_ASYNC_LDS
  __shared__ _Float16 Bl[2][NTILES * 16 * 32];
  const int  bn = tid >> 2;          // local col staged by this thread
  const int  bk = (tid & 3) * 8;     // 8-half k chunk
  const bool bactive = bn < NTILES * 16;
  const _Float16* bsrc = B + (size_t)(n0 + bn) * K + bk;
  if (bactive)
    __builtin_amdgcn_global_load_async_to_lds_b128(
        GPTR(bsrc), LPTR(&Bl[0][bn * 32 + bk]), 0, 0);
  __builtin_amdgcn_s_wait_asynccnt(0);
  __syncthreads();
  int buf = 0;
#endif

  const _Float16* arow = A + (size_t)(m0 + lmod) * K + 8 * lhalf;
  v8h a0 = *(const v8h*)arow;
  v8h a1 = *(const v8h*)(arow + 16);

  for (int kk = 0; kk < K; kk += 32) {
    const int kn = (kk + 32 < K) ? kk + 32 : kk;   // clamped next-tile offset
    v8h a0n = *(const v8h*)(arow + kn);            // pipeline A one iter ahead
    v8h a1n = *(const v8h*)(arow + kn + 16);
    __builtin_prefetch(arow + kn + 32, 0, 1);      // global_prefetch_b8

#if HAVE_ASYNC_LDS
    if ((kk + 32 < K) && bactive)
      __builtin_amdgcn_global_load_async_to_lds_b128(
          GPTR(bsrc + kk + 32), LPTR(&Bl[buf ^ 1][bn * 32 + bk]), 0, 0);
#endif

    // Issue every B-tile load first, then the WMMAs (overlap LDS latency).
    v8h blo[NTILES], bhi[NTILES];
#pragma unroll
    for (int nt = 0; nt < NTILES; ++nt) {
#if HAVE_ASYNC_LDS
      const _Float16* bp = &Bl[buf][(nt * 16 + lmod) * 32 + 8 * lhalf];
#else
      const _Float16* bp = B + (size_t)(n0 + nt * 16 + lmod) * K + kk + 8 * lhalf;
#endif
      blo[nt] = *(const v8h*)bp;
      bhi[nt] = *(const v8h*)(bp + 16);
    }

    v16h a = __builtin_shufflevector(a0, a1, 0,1,2,3,4,5,6,7,8,9,10,11,12,13,14,15);
#pragma unroll
    for (int nt = 0; nt < NTILES; ++nt) {
      v16h b = __builtin_shufflevector(blo[nt], bhi[nt],
                                       0,1,2,3,4,5,6,7,8,9,10,11,12,13,14,15);
      acc[nt] = __builtin_amdgcn_wmma_f32_16x16x32_f16(
          false, a, false, b, (short)0, acc[nt], false, false);
    }
    a0 = a0n; a1 = a1n;

#if HAVE_ASYNC_LDS
    if (kk + 32 < K) {
      __builtin_amdgcn_s_wait_asynccnt(0);
      __syncthreads();
      buf ^= 1;
    }
#endif
  }

#pragma unroll
  for (int nt = 0; nt < NTILES; ++nt) {
    const int col = n0 + nt * 16 + lmod;
    const float bv = bias ? bias[col] : 0.f;
#pragma unroll
    for (int r = 0; r < 8; ++r) {
      const int row = m0 + r + 8 * lhalf;
      C[(size_t)row * N + col] = acc[nt][r] + bv;
    }
  }
}

// ---------------------------------------------------------------------------
// Persistent GRU recurrence. grid = 2 directions * NWG workgroups, 256 threads.
// W_hh (pre-converted to f16 in global) is staged into LDS by the Tensor Data
// Mover: three 32x512 2D tiles (r/z/n gate rows for our 32 hidden units),
// D# groups 0/1 per ISA §8.3/8.4; groups 2/3 (and the trailing int32x8) are
// zero = "unused" for a 2D tile; completion via s_wait_tensorcnt 0.
// ---------------------------------------------------------------------------
__global__ __launch_bounds__(256) void k_gru(
    const _Float16* __restrict__ whh16_f, const float* __restrict__ bhh_f,
    const _Float16* __restrict__ whh16_b, const float* __restrict__ bhh_b,
    const float* __restrict__ gi_f,  const float* __restrict__ gi_b,
    float* __restrict__ hbuf,        // [2 dir][2 phase][512]
    unsigned* __restrict__ sync,     // [2 dir], pre-zeroed
    _Float16* __restrict__ h16)      // [T][1024]  (fwd cols 0..511, bwd 512..1023)
{
  __shared__ _Float16 Wl[96 * 512];  // 96 KB: rows {j, 512+j, 1024+j} of W_hh
  __shared__ float    hs[512];       // full previous hidden state

  const int tid   = threadIdx.x;
  const int dir   = blockIdx.x / NWG;
  const int slice = blockIdx.x % NWG;
  const int j0    = slice * 32;

  const _Float16* whh16 = dir ? whh16_b : whh16_f;
  const float*    bhh   = dir ? bhh_b   : bhh_f;
  const float*    gi    = dir ? gi_b    : gi_f;
  float*    hb = hbuf + dir * 1024;
  unsigned* sc = sync + dir;

#if HAVE_TDM
  if (tid < 32) {   // wave 0 issues three tensor DMAs (one 32x512 tile per gate)
    const unsigned lds_base =
        (unsigned)(unsigned long long)(__attribute__((address_space(3))) const _Float16*)&Wl[0];
#pragma unroll
    for (int g = 0; g < 3; ++g) {
      const unsigned long long ga =
          (unsigned long long)(whh16 + ((size_t)g * 512 + (size_t)j0) * 512);
      // D# group 0: count=1 | lds_addr | global_addr | type=2
      v4u g0 = { 1u,
                 lds_base + (unsigned)(g * 32 * 512 * 2),
                 (unsigned)(ga & 0xFFFFFFFFu),
                 (unsigned)((ga >> 32) & 0x01FFFFFFu) | (2u << 30) };
      // D# group 1: data_size=2B; tensor_dim0=512, tensor_dim1=32;
      //             tile_dim0=512, tile_dim1=32, tile_dim2=0; dim0_stride=512
      v8i g1 = { (int)(1u << 16),
                 (int)(512u << 16),
                 (int)(32u << 16),
                 (int)(512u << 16),
                 32, 512, 0, 0 };
      v4i zz4 = {0, 0, 0, 0};
      v8i zz8 = {0, 0, 0, 0, 0, 0, 0, 0};
      __builtin_amdgcn_tensor_load_to_lds(g0, g1, zz4, zz4, zz8, 0);
    }
    __builtin_amdgcn_s_wait_tensorcnt(0);
  }
#else
  for (int idx = tid; idx < 96 * 512; idx += 256) {
    int lr = idx >> 9, k = idx & 511;
    int row = ((lr >> 5) << 9) + j0 + (lr & 31);   // gate*512 + j
    Wl[idx] = whh16[(size_t)row * 512 + k];
  }
#endif
  hs[tid] = 0.f; hs[tid + 256] = 0.f;              // h0 = 0
  __syncthreads();

  const int jj   = tid >> 3;       // 0..31 local hidden unit
  const int part = tid & 7;        // k-split (8 consecutive lanes)
  const int kb   = part * 64;
  const int j    = j0 + jj;
  const _Float16* wr = &Wl[(jj)      * 512 + kb];
  const _Float16* wz = &Wl[(32 + jj) * 512 + kb];
  const _Float16* wn = &Wl[(64 + jj) * 512 + kb];

  for (int t = 0; t < T_LEN; ++t) {
    float ar = 0.f, az = 0.f, an = 0.f;
#pragma unroll 16
    for (int k = 0; k < 64; k += 2) {
      float h0 = hs[kb + k], h1 = hs[kb + k + 1];
      f16x2 w0 = *(const f16x2*)(wr + k); ar += (float)w0.x * h0 + (float)w0.y * h1;
      f16x2 w1 = *(const f16x2*)(wz + k); az += (float)w1.x * h0 + (float)w1.y * h1;
      f16x2 w2 = *(const f16x2*)(wn + k); an += (float)w2.x * h0 + (float)w2.y * h1;
    }
    // reduce across the 8 k-parts (lanes p..p+7 of each wave32)
    for (int m = 1; m < 8; m <<= 1) {
      ar += __shfl_xor(ar, m);
      az += __shfl_xor(az, m);
      an += __shfl_xor(an, m);
    }

    if (part == 0) {
      const int rowt = dir ? (T_LEN - 1 - t) : t;
      const float* git = gi + (size_t)rowt * G3H;
      float rg = 1.f / (1.f + __expf(-(git[j]        + ar + bhh[j])));
      float zg = 1.f / (1.f + __expf(-(git[512 + j]  + az + bhh[512 + j])));
      float ng = tanhf(git[1024 + j] + rg * (an + bhh[1024 + j]));
      float hn = (1.f - zg) * ng + zg * hs[j];
      hb[((t + 1) & 1) * 512 + j] = hn;
      h16[(size_t)rowt * 1024 + dir * 512 + j] = (_Float16)hn;
    }

    __threadfence();               // make h slice visible at device scope
    __syncthreads();
    if (tid == 0) {
      __hip_atomic_fetch_add(sc, 1u, __ATOMIC_RELEASE, __HIP_MEMORY_SCOPE_AGENT);
      const unsigned target = (unsigned)NWG * (unsigned)(t + 1);
      while (__hip_atomic_load(sc, __ATOMIC_ACQUIRE, __HIP_MEMORY_SCOPE_AGENT) < target)
        __builtin_amdgcn_s_sleep(2);
    }
    __syncthreads();
    // pull the full new hidden state into LDS for the next step
    const float* src = hb + ((t + 1) & 1) * 512;
    hs[tid] = src[tid]; hs[tid + 256] = src[tid + 256];
    __syncthreads();
  }
}

// ---------------------------------------------------------------------------
// Host-side launcher. Workspace layout (all offsets 256B aligned, ~76.7 MB).
// ---------------------------------------------------------------------------
extern "C" void kernel_launch(void* const* d_in, const int* in_sizes, int n_in,
                              void* d_out, int out_size, void* d_ws, size_t ws_size,
                              hipStream_t stream)
{
  (void)in_sizes; (void)n_in; (void)out_size; (void)ws_size;

  const int*   sent  = (const int*)  d_in[0];
  const float* emb   = (const float*)d_in[1];
  const float* Wih_f = (const float*)d_in[2];
  const float* Whh_f = (const float*)d_in[3];
  const float* bih_f = (const float*)d_in[4];
  const float* bhh_f = (const float*)d_in[5];
  const float* Wih_b = (const float*)d_in[6];
  const float* Whh_b = (const float*)d_in[7];
  const float* bih_b = (const float*)d_in[8];
  const float* bhh_b = (const float*)d_in[9];
  const float* Wout  = (const float*)d_in[10];
  float* out = (float*)d_out;

  char* ws = (char*)d_ws;
  _Float16* x16    = (_Float16*)(ws);              //  8,388,608 B  x_f16 [T][E]
  _Float16* wihf16 = (_Float16*)(ws +  8388608);   //  3,145,728 B  W_ih_f f16
  _Float16* wihb16 = (_Float16*)(ws + 11534336);   //  3,145,728 B  W_ih_b f16
  _Float16* wout16 = (_Float16*)(ws + 14680064);   //     65,536 B  W_out f16
  float*    gif    = (float*)   (ws + 14745600);   // 25,165,824 B  gi fwd [T][3H]
  float*    gib    = (float*)   (ws + 39911424);   // 25,165,824 B  gi bwd (natural order)
  _Float16* h16    = (_Float16*)(ws + 65077248);   //  8,388,608 B  h [T][2H] f16
  float*    hbuf   = (float*)   (ws + 73465856);   //      8,192 B  h exchange
  unsigned* syncc  = (unsigned*)(ws + 73474048);   //        256 B  step counters
  _Float16* whh16f = (_Float16*)(ws + 73474304);   //  1,572,864 B  W_hh_f f16
  _Float16* whh16b = (_Float16*)(ws + 75047168);   //  1,572,864 B  W_hh_b f16

  // Phase 1: gather + converts (also zeroes sync counters)
  k_gather<<<dim3(8192), dim3(256), 0, stream>>>(sent, emb, x16, syncc);
  k_cvt<<<dim3(3072), dim3(256), 0, stream>>>(Wih_f, wihf16, G3H * E_DIM);
  k_cvt<<<dim3(3072), dim3(256), 0, stream>>>(Wih_b, wihb16, G3H * E_DIM);
  k_cvt<<<dim3(64),   dim3(256), 0, stream>>>(Wout,  wout16, 32 * E_DIM);
  k_cvt<<<dim3(1536), dim3(256), 0, stream>>>(Whh_f, whh16f, G3H * H_DIM);
  k_cvt<<<dim3(1536), dim3(256), 0, stream>>>(Whh_b, whh16b, G3H * H_DIM);

  // Phase 2: input projections (M=4096, N=1536, K=1024), 16x64 per wave
  k_wmma_gemm<4><<<dim3(G3H / 64, T_LEN / 128), dim3(256), 0, stream>>>(
      x16, wihf16, bih_f, gif, T_LEN, G3H, E_DIM);
  k_wmma_gemm<4><<<dim3(G3H / 64, T_LEN / 128), dim3(256), 0, stream>>>(
      x16, wihb16, bih_b, gib, T_LEN, G3H, E_DIM);

  // Phase 3: persistent bidirectional recurrence (2 x 16 workgroups)
  k_gru<<<dim3(2 * NWG), dim3(256), 0, stream>>>(
      whh16f, bhh_f, whh16b, bhh_b, gif, gib, hbuf, syncc, h16);

  // Phase 4: output projection (M=4096, N=32, K=1024), 16x32 per wave
  k_wmma_gemm<2><<<dim3(1, T_LEN / 128), dim3(256), 0, stream>>>(
      h16, wout16, nullptr, out, T_LEN, 32, E_DIM);
}
